// DigitConvolutionalModel_80779744903987
// MI455X (gfx1250) — matrix operand order repaired and checked by reading, not compiled, not run on hardware
//
#include <hip/hip_runtime.h>

// ---------------------------------------------------------------------------
// DigitConvolutionalModel for MI455X (gfx1250, wave32, WMMA).
// Pipeline: conv3x3(valid) -> [32768 x 676] @ w0^T + b0, relu -> @ w1^T + b1.
// HBM-bound (~200 MB traffic @ 23.3 TB/s ~ 8.5us); bf16 WMMA w/ f32 accum.
// ---------------------------------------------------------------------------

#define BATCH   32768
#define IMGW    28
#define OUTHW   26
#define KCONV   676      // 26*26
#define KPAD    704      // 22 * 32
#define KSTEPS  22
#define NH      200
#define NHPAD   208      // 13 * 16
#define NT      13
#define K2PAD   224      // 7 * 32
#define K2STEPS 7
#define MTILE   128      // 8 waves * 16 rows per block

typedef __attribute__((ext_vector_type(16))) __bf16 v16bf;
typedef __attribute__((ext_vector_type(8)))  float  v8f;

union Frag { v16bf v; uint4 q[2]; };

__device__ __forceinline__ unsigned short f2bf(float f) {
    union { float f; unsigned int u; } c; c.f = f;
    unsigned int u = c.u;
    u += 0x7fffu + ((u >> 16) & 1u);   // round-to-nearest-even
    return (unsigned short)(u >> 16);
}

// ---------------------------------------------------------------------------
// Kernel 1: 3x3 valid conv, fp32 -> bf16 im2row matrix a[BATCH][KPAD].
// One block per sample; x row staged in LDS so x is read exactly once.
// ---------------------------------------------------------------------------
__global__ void conv_bf16(const float* __restrict__ x,
                          const float* __restrict__ cw,
                          unsigned short* __restrict__ a) {
    __shared__ float xs[IMGW * IMGW];
    const int r = blockIdx.x;
    const float* xr = x + (long)r * (IMGW * IMGW);
    for (int i = threadIdx.x; i < IMGW * IMGW; i += blockDim.x) xs[i] = xr[i];

    const float w00 = cw[0], w01 = cw[1], w02 = cw[2];
    const float w10 = cw[3], w11 = cw[4], w12 = cw[5];
    const float w20 = cw[6], w21 = cw[7], w22 = cw[8];
    __syncthreads();

    unsigned short* ar = a + (long)r * KPAD;
    for (int k = threadIdx.x; k < KPAD; k += blockDim.x) {
        float v = 0.0f;
        if (k < KCONV) {
            const int oy = k / OUTHW;
            const int ox = k - oy * OUTHW;
            const float* p = xs + oy * IMGW + ox;
            v = p[0]  * w00 + p[1]  * w01 + p[2]  * w02
              + p[28] * w10 + p[29] * w11 + p[30] * w12
              + p[56] * w20 + p[57] * w21 + p[58] * w22;
        }
        ar[k] = f2bf(v);
    }
}

// ---------------------------------------------------------------------------
// Kernel 2: convert/pad weights to bf16 B-matrix layouts.
//   w0b[NHPAD][KPAD]  (B for GEMM1, column-of-B == contiguous K per row)
//   w1b[16][K2PAD]    (B for GEMM2)
// ---------------------------------------------------------------------------
__global__ void prep_weights(const float* __restrict__ w0,
                             const float* __restrict__ w1,
                             unsigned short* __restrict__ w0b,
                             unsigned short* __restrict__ w1b) {
    const int idx = blockIdx.x * blockDim.x + threadIdx.x;
    if (idx < NHPAD * KPAD) {
        const int n = idx / KPAD;
        const int k = idx - n * KPAD;
        float v = (n < NH && k < KCONV) ? w0[n * KCONV + k] : 0.0f;
        w0b[idx] = f2bf(v);
    } else {
        const int j = idx - NHPAD * KPAD;
        if (j < 16 * K2PAD) {
            const int n = j / K2PAD;
            const int k = j - n * K2PAD;
            float v = (n < 10 && k < NH) ? w1[n * NH + k] : 0.0f;
            w1b[j] = f2bf(v);
        }
    }
}

// ---------------------------------------------------------------------------
// Kernel 3: fused GEMM1 (bf16 WMMA, K=704) + bias/relu + GEMM2 (K=224) + bias.
// 8 waves/block; wave owns 16 rows x all 13 N-tiles (104 accum VGPRs).
// ---------------------------------------------------------------------------
__global__ void __launch_bounds__(256)
fused_gemm(const unsigned short* __restrict__ a,
           const unsigned short* __restrict__ w0b,
           const unsigned short* __restrict__ w1b,
           const float* __restrict__ b0,
           const float* __restrict__ b1,
           float* __restrict__ out) {
    __shared__ __align__(16) unsigned short hbuf[8][16][K2PAD];  // 56 KB

    const int wave  = threadIdx.x >> 5;
    const int lid   = threadIdx.x & 31;
    const int l     = lid & 15;
    const int half  = lid >> 4;
    const int mbase = blockIdx.x * MTILE + wave * 16;

    // zero the K-pad columns [208,224) of this wave's h tile
    for (int idx = lid; idx < 16 * 16; idx += 32)
        hbuf[wave][idx >> 4][NHPAD + (idx & 15)] = 0;

    v8f acc[NT];
#pragma unroll
    for (int t = 0; t < NT; ++t) acc[t] = (v8f){0,0,0,0,0,0,0,0};

    // A fragment source: row (mbase + l), 16-bit 16x32 A layout:
    //   half 0 lanes: K {0..7, 16..23}; half 1 lanes: K {8..15, 24..31}
    const unsigned short* arow = a + (long)(mbase + l) * KPAD;
    const int aoff = 8 * half;

    for (int kk = 0; kk < KSTEPS; ++kk) {
        const int k0 = kk * 32;
        Frag af;
        af.q[0] = *(const uint4*)(arow + k0 + aoff);
        af.q[1] = *(const uint4*)(arow + k0 + 16 + aoff);
        __builtin_prefetch(arow + k0 + 64, 0, 1);   // global_prefetch_b8
#pragma unroll
        for (int t = 0; t < NT; ++t) {
            // B fragment: column n = t*16 + l, K {8h..8h+7, 16+8h..16+8h+7}
            const unsigned short* brow = w0b + (long)(t * 16 + l) * KPAD + k0 + 8 * half;
            Frag bf;
            bf.q[0] = *(const uint4*)(brow);
            bf.q[1] = *(const uint4*)(brow + 16);
            acc[t] = __builtin_amdgcn_wmma_f32_16x16x32_bf16(
                false, af.v, false, bf.v, (short)0, acc[t], false, false);
        }
    }

    // bias + relu, stash h tile as bf16 in per-wave LDS
#pragma unroll
    for (int t = 0; t < NT; ++t) {
        const int col  = t * 16 + l;
        const int bidx = col < NH ? col : 0;
        float bias = b0[bidx];
        if (col >= NH) bias = 0.0f;
#pragma unroll
        for (int i = 0; i < 8; ++i) {
            float hv = acc[t][i] + bias;
            hv = hv > 0.0f ? hv : 0.0f;
            hbuf[wave][i + 8 * half][col] = f2bf(hv);
        }
    }

    // GEMM2: (16 x 224) @ (224 x 16), only 10 output columns valid
    v8f acc2 = (v8f){0,0,0,0,0,0,0,0};
#pragma unroll
    for (int kk = 0; kk < K2STEPS; ++kk) {
        const int k0 = kk * 32;
        Frag af;
        af.q[0] = *(const uint4*)&hbuf[wave][l][k0 + aoff];
        af.q[1] = *(const uint4*)&hbuf[wave][l][k0 + 16 + aoff];
        const unsigned short* brow = w1b + (long)l * K2PAD + k0 + 8 * half;
        Frag bf;
        bf.q[0] = *(const uint4*)(brow);
        bf.q[1] = *(const uint4*)(brow + 16);
        acc2 = __builtin_amdgcn_wmma_f32_16x16x32_bf16(
            false, af.v, false, bf.v, (short)0, acc2, false, false);
    }

    if (l < 10) {
        const float bias1 = b1[l];
#pragma unroll
        for (int i = 0; i < 8; ++i) {
            const int row = mbase + i + 8 * half;
            out[(long)row * 10 + l] = acc2[i] + bias1;
        }
    }
}

// ---------------------------------------------------------------------------
extern "C" void kernel_launch(void* const* d_in, const int* in_sizes, int n_in,
                              void* d_out, int out_size, void* d_ws, size_t ws_size,
                              hipStream_t stream) {
    const float* x  = (const float*)d_in[0];
    const float* cw = (const float*)d_in[1];
    const float* w0 = (const float*)d_in[2];
    const float* b0 = (const float*)d_in[3];
    const float* w1 = (const float*)d_in[4];
    const float* b1 = (const float*)d_in[5];
    float* out = (float*)d_out;

    unsigned short* a   = (unsigned short*)d_ws;            // BATCH*KPAD   (46.1 MB)
    unsigned short* w0b = a + (size_t)BATCH * KPAD;         // NHPAD*KPAD   (0.29 MB)
    unsigned short* w1b = w0b + (size_t)NHPAD * KPAD;       // 16*K2PAD     (7 KB)

    conv_bf16<<<BATCH, 256, 0, stream>>>(x, cw, a);

    const int prep_elems = NHPAD * KPAD + 16 * K2PAD;
    prep_weights<<<(prep_elems + 255) / 256, 256, 0, stream>>>(w0, w1, w0b, w1b);

    fused_gemm<<<BATCH / MTILE, 256, 0, stream>>>(a, w0b, w1b, b0, b1, out);
}
